// flow_mapping_10728828305559
// MI455X (gfx1250) — matrix operand order repaired and checked by reading, not compiled, not run on hardware
//
#include <hip/hip_runtime.h>
#include <hip/hip_bf16.h>
#include <math.h>

typedef __attribute__((ext_vector_type(16))) _Float16 v16h;
typedef __attribute__((ext_vector_type(8)))  _Float16 v8h;
typedef __attribute__((ext_vector_type(8)))  float    v8f;

#define BATCH   262144
#define NDIM    8
#define NSPLIT  4
#define DEPTH_  8
#define WIDTH_  32
#define NOUT    8
#define ALPHA_  0.6f
#define RBLOCKS 1024
#define TBLOCKS 512

// ---------------------------------------------------------------------------
// Stage 1: per-block partial sum / sum-of-squares per feature (deterministic
// fixed-tree reduction, no float atomics).
// ---------------------------------------------------------------------------
__global__ void __launch_bounds__(256)
reduce_stage1(const float* __restrict__ z, float* __restrict__ partials)
{
    __shared__ float rs[256];
    __shared__ float rq[256];
    const int t = threadIdx.x;
    const int f = t & 7;
    long m0 = (long)blockIdx.x * 32 + (t >> 3);
    const long stride = (long)gridDim.x * 32;
    float s = 0.f, q = 0.f;
    for (long m = m0; m < BATCH; m += stride) {
        float v = z[m * NDIM + f];
        s += v; q += v * v;
    }
    rs[t] = s; rq[t] = q;
    __syncthreads();
    for (int off = 128; off >= 8; off >>= 1) {
        if (t < off) { rs[t] += rs[t + off]; rq[t] += rq[t + off]; }
        __syncthreads();
    }
    if (t < 8) {
        partials[blockIdx.x * 16 + t]     = rs[t];
        partials[blockIdx.x * 16 + 8 + t] = rq[t];
    }
}

// Stage 2: serial (fixed-order, deterministic) reduction of block partials.
__global__ void reduce_stage2(const float* __restrict__ partials,
                              float* __restrict__ stats)
{
    const int t = threadIdx.x;
    if (t < 16) {
        float s = 0.f;
        for (int b = 0; b < RBLOCKS; ++b) s += partials[b * 16 + t];
        stats[t] = s;
    }
}

// ---------------------------------------------------------------------------
// Fused actnorm + 3-layer MLP (WMMA f16, f32 accum) + affine coupling + logdet
// + feature reversal, for one flow layer. Each wave owns a 32-sample tile
// (two 16-row WMMA M-tiles) so all 32 lanes carry one sample through the
// elementwise phases; D->A transposes between GEMMs go through a private
// per-wave LDS slice (two 16B ds_loads land exactly in the f16 A layout).
// ---------------------------------------------------------------------------
__global__ void __launch_bounds__(256)
flow_layer(const float* __restrict__ zin,  float* __restrict__ zout,
           const float* __restrict__ ldin, float* __restrict__ ldout,
           const float* __restrict__ stats,
           const float* __restrict__ bP,  const float* __restrict__ logsP,
           const float* __restrict__ lgP,
           const float* __restrict__ W1P, const float* __restrict__ b1P,
           const float* __restrict__ W2P, const float* __restrict__ b2P,
           const float* __restrict__ WfP, const float* __restrict__ bfP,
           int layer)
{
    __shared__ __align__(16) _Float16 hbuf[8][32 * 32];  // per-wave h tiles (f16)
    __shared__ __align__(16) float    h3buf[8][32 * 8];  // per-wave final MLP out

    const int tid  = threadIdx.x;
    const int w    = tid >> 5;    // wave id in block
    const int lane = tid & 31;
    const int n0   = lane & 15;   // WMMA column / row-within-M-tile
    const int half = lane >> 4;   // lane half (WMMA layout split)
    const int mB   = half * 8;    // D-tile row base for this lane half

    const float* b_l    = bP    + layer * NDIM;
    const float* logs_l = logsP + layer * NDIM;
    const float* lg_l   = lgP   + layer * NSPLIT;
    const float* W1     = W1P   + layer * NSPLIT * WIDTH_;
    const float* b1     = b1P   + layer * WIDTH_;
    const float* W2     = W2P   + layer * WIDTH_ * WIDTH_;
    const float* b2     = b2P   + layer * WIDTH_;
    const float* Wf     = WfP   + layer * WIDTH_ * NOUT;
    const float* bf     = bfP   + layer * NOUT;

    // Actnorm constants from batch stats (uniform across all threads).
    float bb[NDIM], es[NDIM], sum_s = 0.f;
    #pragma unroll
    for (int f = 0; f < NDIM; ++f) {
        float mean = stats[f]     * (1.f / BATCH);
        float ex2  = stats[8 + f] * (1.f / BATCH);
        float var  = fmaxf(ex2 - mean * mean, 0.f);
        float logs_init = -logf(sqrtf(var) + 1e-6f) * (1.f / 3.0f);
        float s = fminf(fmaxf(logs_l[f] + logs_init, -5.f), 5.f);
        es[f] = expf(s);
        bb[f] = b_l[f] - mean;
        sum_s += s;
    }
    float eg[NSPLIT];
    #pragma unroll
    for (int f = 0; f < NSPLIT; ++f)
        eg[f] = expf(fminf(fmaxf(lg_l[f], -5.f), 5.f));

    // Weights -> f16 B-matrix register tiles (32x16 K-major WMMA layout:
    // lanes 0-15 hold K=0-7 & 16-23, lanes 16-31 hold K=8-15 & 24-31, N = lane%16).
    const int kb = half * 8;
    v16h B1t[2], B2t[2], B3t;
    #pragma unroll
    for (int j = 0; j < 16; ++j) {
        int k = (j < 8) ? (kb + j) : (16 + kb + (j - 8));
        B1t[0][j] = (k < NSPLIT) ? (_Float16)W1[k * WIDTH_ + n0]        : (_Float16)0.f;
        B1t[1][j] = (k < NSPLIT) ? (_Float16)W1[k * WIDTH_ + 16 + n0]   : (_Float16)0.f;
        B2t[0][j] = (_Float16)W2[k * WIDTH_ + n0];
        B2t[1][j] = (_Float16)W2[k * WIDTH_ + 16 + n0];
        B3t[j]    = (n0 < NOUT) ? (_Float16)Wf[k * NOUT + n0] : (_Float16)0.f;
    }
    const float b1n0 = b1[n0],      b1n1 = b1[16 + n0];
    const float b2n0 = b2[n0],      b2n1 = b2[16 + n0];
    const float bfn  = (n0 < NOUT) ? bf[n0] : 0.f;

    const _Float16 h0 = (_Float16)0.f;
    const int ntiles = BATCH / 32;
    for (int T = blockIdx.x * 8 + w; T < ntiles; T += gridDim.x * 8) {
        const long gs = (long)T * 32 + lane;   // every lane owns one sample
        float zp[NDIM];
        #pragma unroll
        for (int f = 0; f < NDIM; ++f)
            zp[f] = (zin[gs * NDIM + f] + bb[f]) * es[f];

        // ---- GEMM1: (32x4) @ (4x32) as two M-tiles, K padded to 32.
        // M-tile 0 = samples T*32+0..15 (lanes 0-15 own them directly);
        // M-tile 1 = samples T*32+16..31 (shuffled down from lanes 16-31).
        v16h aM0 = {}, aM1 = {};
        #pragma unroll
        for (int j = 0; j < NSPLIT; ++j) {
            float up = __shfl(zp[j], (lane & 15) + 16, 32);
            aM0[j] = (half == 0) ? (_Float16)zp[j] : h0;
            aM1[j] = (half == 0) ? (_Float16)up    : h0;
        }
        v8f z8 = {};
        v8f c00 = __builtin_amdgcn_wmma_f32_16x16x32_f16(false, aM0, false, B1t[0], (short)0, z8, false, false);
        v8f c01 = __builtin_amdgcn_wmma_f32_16x16x32_f16(false, aM0, false, B1t[1], (short)0, z8, false, false);
        v8f c10 = __builtin_amdgcn_wmma_f32_16x16x32_f16(false, aM1, false, B1t[0], (short)0, z8, false, false);
        v8f c11 = __builtin_amdgcn_wmma_f32_16x16x32_f16(false, aM1, false, B1t[1], (short)0, z8, false, false);

        // bias + relu; stage D tiles into LDS as h[sample][hidden] f16.
        #pragma unroll
        for (int r = 0; r < 8; ++r) {
            hbuf[w][(mB + r) * 32 + n0]           = (_Float16)fmaxf(c00[r] + b1n0, 0.f);
            hbuf[w][(mB + r) * 32 + 16 + n0]      = (_Float16)fmaxf(c01[r] + b1n1, 0.f);
            hbuf[w][(16 + mB + r) * 32 + n0]      = (_Float16)fmaxf(c10[r] + b1n0, 0.f);
            hbuf[w][(16 + mB + r) * 32 + 16 + n0] = (_Float16)fmaxf(c11[r] + b1n1, 0.f);
        }
        __builtin_amdgcn_wave_barrier();

        // Reload in A layout (per M-tile): lane<16 -> M=lane, K 0-7 & 16-23;
        // lane>=16 -> M=lane-16, K 8-15 & 24-31. Two 16-byte LDS loads each.
        {
            const _Float16* hr0 = &hbuf[w][n0 * 32 + half * 8];
            const _Float16* hr1 = &hbuf[w][(16 + n0) * 32 + half * 8];
            v8h lo0 = *(const v8h*)(hr0);
            v8h hi0 = *(const v8h*)(hr0 + 16);
            v8h lo1 = *(const v8h*)(hr1);
            v8h hi1 = *(const v8h*)(hr1 + 16);
            #pragma unroll
            for (int j = 0; j < 8; ++j) {
                aM0[j] = lo0[j]; aM0[8 + j] = hi0[j];
                aM1[j] = lo1[j]; aM1[8 + j] = hi1[j];
            }
        }

        // ---- GEMM2: (32x32) @ (32x32); K=32 in one WMMA per tile.
        v8f d00 = __builtin_amdgcn_wmma_f32_16x16x32_f16(false, aM0, false, B2t[0], (short)0, z8, false, false);
        v8f d01 = __builtin_amdgcn_wmma_f32_16x16x32_f16(false, aM0, false, B2t[1], (short)0, z8, false, false);
        v8f d10 = __builtin_amdgcn_wmma_f32_16x16x32_f16(false, aM1, false, B2t[0], (short)0, z8, false, false);
        v8f d11 = __builtin_amdgcn_wmma_f32_16x16x32_f16(false, aM1, false, B2t[1], (short)0, z8, false, false);
        #pragma unroll
        for (int r = 0; r < 8; ++r) {
            hbuf[w][(mB + r) * 32 + n0]           = (_Float16)fmaxf(d00[r] + b2n0, 0.f);
            hbuf[w][(mB + r) * 32 + 16 + n0]      = (_Float16)fmaxf(d01[r] + b2n1, 0.f);
            hbuf[w][(16 + mB + r) * 32 + n0]      = (_Float16)fmaxf(d10[r] + b2n0, 0.f);
            hbuf[w][(16 + mB + r) * 32 + 16 + n0] = (_Float16)fmaxf(d11[r] + b2n1, 0.f);
        }
        __builtin_amdgcn_wave_barrier();
        {
            const _Float16* hr0 = &hbuf[w][n0 * 32 + half * 8];
            const _Float16* hr1 = &hbuf[w][(16 + n0) * 32 + half * 8];
            v8h lo0 = *(const v8h*)(hr0);
            v8h hi0 = *(const v8h*)(hr0 + 16);
            v8h lo1 = *(const v8h*)(hr1);
            v8h hi1 = *(const v8h*)(hr1 + 16);
            #pragma unroll
            for (int j = 0; j < 8; ++j) {
                aM0[j] = lo0[j]; aM0[8 + j] = hi0[j];
                aM1[j] = lo1[j]; aM1[8 + j] = hi1[j];
            }
        }

        // ---- GEMM3: (32x32) @ (32x8), N padded to 16.
        v8f e0 = __builtin_amdgcn_wmma_f32_16x16x32_f16(false, aM0, false, B3t, (short)0, z8, false, false);
        v8f e1 = __builtin_amdgcn_wmma_f32_16x16x32_f16(false, aM1, false, B3t, (short)0, z8, false, false);
        if (n0 < NOUT) {
            #pragma unroll
            for (int r = 0; r < 8; ++r) {
                h3buf[w][(mB + r) * 8 + n0]        = e0[r] + bfn;
                h3buf[w][(16 + mB + r) * 8 + n0]   = e1[r] + bfn;
            }
        }
        __builtin_amdgcn_wave_barrier();

        // ---- Coupling + logdet + reversal (all 32 lanes, one sample each).
        {
            const float* hp = &h3buf[w][lane * 8];
            float full[NDIM];
            float ldacc = 0.f;
            #pragma unroll
            for (int f = 0; f < NSPLIT; ++f) full[f] = zp[f];
            #pragma unroll
            for (int f = 0; f < NSPLIT; ++f) {
                float sc = ALPHA_ * tanhf(hp[2 * f + 1]);
                float sh = eg[f]  * tanhf(hp[2 * f]);
                full[NSPLIT + f] = zp[NSPLIT + f] * (1.f + sc) + sh;
                ldacc += log1pf(sc);
            }
            float4 r0 = make_float4(full[7], full[6], full[5], full[4]);
            float4 r1 = make_float4(full[3], full[2], full[1], full[0]);
            *(float4*)(&zout[gs * NDIM])     = r0;  // z = concat(z1,z2)[::-1]
            *(float4*)(&zout[gs * NDIM + 4]) = r1;
            ldout[gs] = ldin[gs] + sum_s + ldacc;
        }
        __builtin_amdgcn_wave_barrier();
    }
}

// ---------------------------------------------------------------------------
extern "C" void kernel_launch(void* const* d_in, const int* in_sizes, int n_in,
                              void* d_out, int out_size, void* d_ws, size_t ws_size,
                              hipStream_t stream)
{
    const float* x     = (const float*)d_in[0];
    const float* ld0   = (const float*)d_in[1];
    const float* bP    = (const float*)d_in[2];
    const float* logsP = (const float*)d_in[3];
    const float* lgP   = (const float*)d_in[4];
    const float* W1P   = (const float*)d_in[5];
    const float* b1P   = (const float*)d_in[6];
    const float* W2P   = (const float*)d_in[7];
    const float* b2P   = (const float*)d_in[8];
    const float* WfP   = (const float*)d_in[9];
    const float* bfP   = (const float*)d_in[10];

    float* out_z  = (float*)d_out;                // B*8 floats
    float* out_ld = out_z + (size_t)BATCH * NDIM; // B floats

    float* ws_z     = (float*)d_ws;                  // B*8 floats
    float* ws_ld    = ws_z + (size_t)BATCH * NDIM;   // B floats
    float* partials = ws_ld + BATCH;                 // RBLOCKS*16 floats
    float* stats    = partials + RBLOCKS * 16;       // 16 floats
    // total scratch ~9.4 MB

    for (int layer = 0; layer < DEPTH_; ++layer) {
        const float* zin    = (layer == 0) ? x   : ws_z;
        float*       zoutp  = (layer == DEPTH_ - 1) ? out_z  : ws_z;
        const float* ldin   = (layer == 0) ? ld0 : ws_ld;
        float*       ldoutp = (layer == DEPTH_ - 1) ? out_ld : ws_ld;

        reduce_stage1<<<RBLOCKS, 256, 0, stream>>>(zin, partials);
        reduce_stage2<<<1, 32, 0, stream>>>(partials, stats);
        flow_layer<<<TBLOCKS, 256, 0, stream>>>(zin, zoutp, ldin, ldoutp, stats,
                                                bP, logsP, lgP, W1P, b1P,
                                                W2P, b2P, WfP, bfP, layer);
    }
}
